// RSA_12214886990346
// MI455X (gfx1250) — matrix operand order repaired and checked by reading, not compiled, not run on hardware
//
#include <hip/hip_runtime.h>
#include <hip/hip_bf16.h>

// ---------------------------------------------------------------------------
// MI455X / gfx1250.  roll(-4)+qkv GEMM (WMMA bf16) -> depthwise 3x3 ->
// windowed channel attention (2x WMMA GEMMs/window) -> roll(+4)+proj GEMM.
// bf16 intermediates in d_ws.  WMMA fragments are contiguous 16B LDS vector
// loads (ds_load_b128) with k = 16*half + s in each lane (A and B alike); the 1x1-conv
// GEMMs double-buffer their LDS tiles so global staging of tile k+1 overlaps
// the WMMAs of tile k.
// ---------------------------------------------------------------------------

typedef __attribute__((ext_vector_type(16))) __bf16 v16bf;
typedef __attribute__((ext_vector_type(8)))  __bf16 v8bf;
typedef __attribute__((ext_vector_type(8)))  float  v8f;

__device__ __forceinline__ __bf16 f2bf(float f) {
  unsigned u = __builtin_bit_cast(unsigned, f);
  unsigned r = (u + 0x7FFFu + ((u >> 16) & 1u)) >> 16;   // RNE
  unsigned short s = (unsigned short)r;
  return __builtin_bit_cast(__bf16, s);
}
__device__ __forceinline__ float bf2f(__bf16 b) {
  unsigned short s = __builtin_bit_cast(unsigned short, b);
  unsigned u = ((unsigned)s) << 16;
  return __builtin_bit_cast(float, u);
}

// 16-element fragment from two contiguous 16B LDS runs.
__device__ __forceinline__ v16bf frag2(const __bf16* lo, const __bf16* hi) {
  union { v16bf v; v8bf h[2]; } u;
  u.h[0] = *(const v8bf*)lo;
  u.h[1] = *(const v8bf*)hi;
  return u.v;
}
// 16-element fragment from one contiguous 32B LDS run.
__device__ __forceinline__ v16bf frag1(const __bf16* p) {
  union { v16bf v; v8bf h[2]; } u;
  u.h[0] = *(const v8bf*)p;
  u.h[1] = *(const v8bf*)(p + 8);
  return u.v;
}

// ---------------------------------------------------------------------------
// Fused roll + 1x1 conv:  Y[o,p] = sum_c W[o,c] * X[c, roll(p)]
// Block 256 thr (8 waves), tile BM=64 x BN=128 x BK=32, double-buffered LDS.
// Wave (wm 0..3, wn 0..1): rows wm*16..+15, cols wn*64..+63 (4 N-tiles).
// ---------------------------------------------------------------------------
template <int MTOT, int ROLL, bool IN_BF16, bool OUT_F32>
__global__ __launch_bounds__(256) void conv1x1_wmma(
    const float* __restrict__ Wmat,   // [MTOT, 256] f32
    const void*  __restrict__ Xin,    // [b, 256, 256, 256]
    void*        __restrict__ Yout)   // [b, MTOT, 256, 256]
{
  constexpr int BM = 64, BN = 128, BK = 32, NKT = 256 / BK;
  __shared__ __attribute__((aligned(32))) __bf16 As[2][BM][BK + 8];  // row-major A
  __shared__ __attribute__((aligned(32))) __bf16 Bt[2][BN][BK + 8];  // transposed B

  const int tid  = threadIdx.x;
  const int lane = tid & 31, wid = tid >> 5;
  const int half = lane >> 4, l16 = lane & 15;
  const int wm = wid & 3, wn = wid >> 2;
  const int m0 = blockIdx.y * BM;
  const long long n0 = (long long)blockIdx.x * BN;

  const float*  Xf = (const float*)Xin;
  const __bf16* Xb = (const __bf16*)Xin;

  // stage tile `kt` into buffer `buf`
  auto stage = [&](int kt, int buf) {
    for (int i = tid; i < BM * BK; i += 256) {      // A (row major)
      int m = i >> 5, k = i & 31;
      As[buf][m][k] = f2bf(Wmat[(m0 + m) * 256 + kt * BK + k]);
    }
    for (int i = tid; i < BK * BN; i += 256) {      // B transposed
      int k = i >> 7, n = i & 127;                  // n fastest -> coalesced
      long long p = n0 + n;
      int batch = (int)(p >> 16);
      int rem   = (int)(p & 65535);
      int h = rem >> 8, w = rem & 255;
      int hs = (h + ROLL) & 255, ws2 = (w + ROLL) & 255;
      long long src = (((long long)(batch * 256 + kt * BK + k)) << 16) + (hs << 8) + ws2;
      Bt[buf][n][k] = IN_BF16 ? Xb[src] : f2bf(Xf[src]);
    }
  };

  v8f acc[4];
  #pragma unroll
  for (int i = 0; i < 4; ++i)
    #pragma unroll
    for (int r = 0; r < 8; ++r) acc[i][r] = 0.0f;

  stage(0, 0);
  __syncthreads();

  for (int kt = 0; kt < NKT; ++kt) {
    const int cur = kt & 1;
    if (kt + 1 < NKT) stage(kt + 1, cur ^ 1);  // overlaps with WMMAs below

    const __bf16* arow = &As[cur][wm * 16 + l16][0];
    v16bf a = frag1(arow + half * 16);

    #pragma unroll
    for (int nt = 0; nt < 4; ++nt) {
      const __bf16* brow = &Bt[cur][wn * 64 + nt * 16 + l16][0];
      v16bf b = frag1(brow + half * 16);
      acc[nt] = __builtin_amdgcn_wmma_f32_16x16x32_bf16(
          false, a, false, b, (short)0, acc[nt], false, false);
    }
    __syncthreads();   // next tile staged & this tile's reads done
  }

  #pragma unroll
  for (int nt = 0; nt < 4; ++nt) {
    #pragma unroll
    for (int r = 0; r < 8; ++r) {
      int o = m0 + wm * 16 + half * 8 + r;
      long long p = n0 + wn * 64 + nt * 16 + l16;
      int batch = (int)(p >> 16);
      int rem   = (int)(p & 65535);
      long long dst = (((long long)(batch * MTOT + o)) << 16) + rem;
      float v = acc[nt][r];
      if constexpr (OUT_F32) ((float*)Yout)[dst] = v;
      else                   ((__bf16*)Yout)[dst] = f2bf(v);
    }
  }
}

// ---------------------------------------------------------------------------
// Depthwise 3x3, groups = 768, zero padding. bf16 in/out, f32 weights.
// ---------------------------------------------------------------------------
__global__ __launch_bounds__(256) void dwconv3x3(
    const __bf16* __restrict__ in,   // [4, 768, 256, 256]
    const float*  __restrict__ wdw,  // [768, 1, 3, 3]
    __bf16*       __restrict__ out)
{
  long long idx = (long long)blockIdx.x * 256 + threadIdx.x;
  int w = (int)(idx & 255);
  int h = (int)((idx >> 8) & 255);
  int ch = (int)((idx >> 16) % 768);
  long long base = idx - (h << 8) - w;
  const float* wt = wdw + ch * 9;
  float acc = 0.0f;
  #pragma unroll
  for (int dy = -1; dy <= 1; ++dy) {
    #pragma unroll
    for (int dx = -1; dx <= 1; ++dx) {
      int hh = h + dy, ww = w + dx;
      if (hh >= 0 && hh < 256 && ww >= 0 && ww < 256)
        acc += wt[(dy + 1) * 3 + (dx + 1)] * bf2f(in[base + (hh << 8) + ww]);
    }
  }
  out[idx] = f2bf(acc);
}

// ---------------------------------------------------------------------------
// Windowed channel attention. One block (8 waves) per 8x8 window.
// LDS layouts give 16B-contiguous fragments:
//   Qt[ch][px], Kt[ch][px] : channel-major (GEMM1 A = q^T rows, B = k cols)
//   Vs[px][ch]             : pixel-major   (GEMM2 A rows)
//   AtnT[d][c]             : transposed attn chunk (GEMM2 B rows)
// LDS: 2*36864 + 33792 + 16896 + 512 = 124,928 B (< 320 KB WGP LDS)
// ---------------------------------------------------------------------------
__global__ __launch_bounds__(256) void winattn_wmma(
    const __bf16* __restrict__ qkv,   // [4, 768, 256, 256] (shifted space)
    const float*  __restrict__ temp,  // [1]
    __bf16*       __restrict__ outb)  // [4, 256, 256, 256] (shifted space)
{
  __shared__ __attribute__((aligned(32))) __bf16 Qt[256][72];
  __shared__ __attribute__((aligned(32))) __bf16 Kt[256][72];
  __shared__ __attribute__((aligned(32))) __bf16 Vs[64][264];
  __shared__ __attribute__((aligned(32))) __bf16 AtnT[32][264];
  __shared__ float scl[128];

  const int tid  = threadIdx.x;
  const int lane = tid & 31, wid = tid >> 5;
  const int half = lane >> 4, l16 = lane & 15;

  const int wi    = blockIdx.x;          // 0..4095
  const int batch = wi >> 10;
  const int wrem  = wi & 1023;
  const int wh = wrem >> 5, ww = wrem & 31;

  // Load q,k,v windows (p fastest: 8-contiguous global reads per row)
  for (int i = tid; i < 64 * 256; i += 256) {
    int ch = i >> 6, p = i & 63;
    int h = wh * 8 + (p >> 3), w = ww * 8 + (p & 7);
    long long pix = (((long long)batch * 768) << 16) + (h << 8) + w;
    Qt[ch][p] = qkv[pix + (((long long)ch) << 16)];
    Kt[ch][p] = qkv[pix + (((long long)(ch + 256)) << 16)];
    Vs[p][ch] = qkv[pix + (((long long)(ch + 512)) << 16)];
  }
  __syncthreads();

  // Per-pixel L2 norms along channels (threads 0-63: q, 64-127: k)
  if (tid < 128) {
    int p = tid & 63;
    float s = 0.0f;
    if (tid < 64) { for (int c = 0; c < 256; ++c) { float v = bf2f(Qt[c][p]); s += v * v; } }
    else          { for (int c = 0; c < 256; ++c) { float v = bf2f(Kt[c][p]); s += v * v; } }
    scl[tid] = 1.0f / fmaxf(sqrtf(s), 1e-12f);
  }
  __syncthreads();
  for (int i = tid; i < 64 * 256; i += 256) {
    int c = i >> 6, p = i & 63;
    Qt[c][p] = f2bf(bf2f(Qt[c][p]) * scl[p]);
    Kt[c][p] = f2bf(bf2f(Kt[c][p]) * scl[64 + p]);
  }
  __syncthreads();

  const float tau = temp[0];

  for (int dc = 0; dc < 8; ++dc) {
    // ---- GEMM1: attn[c,d] = sum_p Qn[p,c]*Kn[p,d]  (M=256,N=32,K=64) ----
    v8f acc1[2][2];
    #pragma unroll
    for (int mi = 0; mi < 2; ++mi)
      #pragma unroll
      for (int ni = 0; ni < 2; ++ni)
        #pragma unroll
        for (int r = 0; r < 8; ++r) acc1[mi][ni][r] = 0.0f;

    #pragma unroll
    for (int kt = 0; kt < 2; ++kt) {
      v16bf afr[2], bfr[2];
      #pragma unroll
      for (int mi = 0; mi < 2; ++mi) {
        const __bf16* q = &Qt[wid * 32 + mi * 16 + l16][kt * 32];
        afr[mi] = frag1(q + half * 16);
      }
      #pragma unroll
      for (int ni = 0; ni < 2; ++ni) {
        const __bf16* k = &Kt[dc * 32 + ni * 16 + l16][kt * 32];
        bfr[ni] = frag1(k + half * 16);
      }
      #pragma unroll
      for (int mi = 0; mi < 2; ++mi)
        #pragma unroll
        for (int ni = 0; ni < 2; ++ni)
          acc1[mi][ni] = __builtin_amdgcn_wmma_f32_16x16x32_bf16(
              false, afr[mi], false, bfr[ni], (short)0, acc1[mi][ni],
              false, false);
    }

    // relu(tau * attn) -> AtnT[d][c]  (single v_max_f32 per element,
    // 8 consecutive bf16 per lane -> mergeable ds_store)
    #pragma unroll
    for (int mi = 0; mi < 2; ++mi)
      #pragma unroll
      for (int ni = 0; ni < 2; ++ni) {
        int d = ni * 16 + l16;
        int c0 = wid * 32 + mi * 16 + half * 8;
        #pragma unroll
        for (int r = 0; r < 8; ++r)
          AtnT[d][c0 + r] = f2bf(fmaxf(tau * acc1[mi][ni][r], 0.0f));
      }
    __syncthreads();

    // ---- GEMM2: out[p,d] = sum_c V[p,c]*attn[c,d]  (M=64,N=32,K=256) ----
    const int wm = wid & 3, wn = wid >> 2;
    v8f acc2;
    #pragma unroll
    for (int r = 0; r < 8; ++r) acc2[r] = 0.0f;

    #pragma unroll
    for (int kt = 0; kt < 8; ++kt) {
      const __bf16* v = &Vs[wm * 16 + l16][kt * 32];
      v16bf afr = frag1(v + half * 16);
      const __bf16* at = &AtnT[wn * 16 + l16][kt * 32];
      v16bf bfr = frag1(at + half * 16);
      acc2 = __builtin_amdgcn_wmma_f32_16x16x32_bf16(
          false, afr, false, bfr, (short)0, acc2, false, false);
    }

    #pragma unroll
    for (int r = 0; r < 8; ++r) {
      int p = wm * 16 + half * 8 + r;
      int d = dc * 32 + wn * 16 + l16;
      int h = wh * 8 + (p >> 3), w = ww * 8 + (p & 7);
      long long dst = (((long long)(batch * 256 + d)) << 16) + (h << 8) + w;
      outb[dst] = f2bf(acc2[r]);
    }
    __syncthreads();   // AtnT reused next chunk
  }
}

// ---------------------------------------------------------------------------
// Launcher.  Workspace (bf16): qkv 402653184 B | qkv_dw 402653184 B |
//                              attn_out 134217728 B
// ---------------------------------------------------------------------------
extern "C" void kernel_launch(void* const* d_in, const int* in_sizes, int n_in,
                              void* d_out, int out_size, void* d_ws, size_t ws_size,
                              hipStream_t stream) {
  const float* x      = (const float*)d_in[0];  // [4,256,256,256]
  const float* w_qkv  = (const float*)d_in[1];  // [768,256]
  const float* w_dw   = (const float*)d_in[2];  // [768,1,3,3]
  const float* w_proj = (const float*)d_in[3];  // [256,256]
  const float* temp   = (const float*)d_in[4];  // [1]
  float* y = (float*)d_out;                     // [4,256,256,256]

  char* ws = (char*)d_ws;
  __bf16* qkv      = (__bf16*)(ws);
  __bf16* qkv_dw   = (__bf16*)(ws + 402653184LL);
  __bf16* attn_out = (__bf16*)(ws + 805306368LL);

  dim3 blk(256);

  // 1) qkv = w_qkv @ roll(x, -4)
  conv1x1_wmma<768, 4, false, false>
      <<<dim3(8192, 12), blk, 0, stream>>>(w_qkv, (const void*)x, (void*)qkv);

  // 2) depthwise 3x3
  dwconv3x3<<<dim3(786432), blk, 0, stream>>>(qkv, w_dw, qkv_dw);

  // 3) windowed channel attention
  winattn_wmma<<<dim3(4096), blk, 0, stream>>>(qkv_dw, temp, attn_out);

  // 4) y = w_proj @ roll(out, +4)  (f32 out)
  conv1x1_wmma<256, 252, true, true>
      <<<dim3(8192, 4), blk, 0, stream>>>(w_proj, (const void*)attn_out, (void*)y);
}
